// VisionMambaBlock_42322607735353
// MI455X (gfx1250) — compile-verified, matrix operand order checked by reading, not run on hardware
//
#include <hip/hip_runtime.h>
#include <hip/hip_bf16.h>
#include <math.h>

typedef __attribute__((ext_vector_type(16))) _Float16 v16h;
typedef __attribute__((ext_vector_type(8)))  float    v8f;

#define D_MODEL 512
#define D_INNER 1024
#define D_STATE 16
#define DT_RANK 32
#define SEQ_B   8
#define SEQ_L   2048
#define NTOK    (SEQ_B * SEQ_L)      // 16384 tokens

#define BM 128                        // block tile M
#define BN 128                        // block tile N
#define LDS_STRIDE 20                 // dwords per row: 16 data + 4 pad (conflict-free, 16B aligned)

union Frag { v16h h; unsigned u[8]; };

__device__ __forceinline__ float fast_silu(float x) {
    return x * (1.0f / (1.0f + __expf(-x)));
}

// ---------------------------------------------------------------------------
// f32 -> f16 conversion (weights + activations)
// ---------------------------------------------------------------------------
__global__ void cvt_f32_to_f16(const float* __restrict__ in,
                               _Float16* __restrict__ out, int n) {
    int i = blockIdx.x * 256 + threadIdx.x;
    if (i < n) out[i] = (_Float16)in[i];
}

// ---------------------------------------------------------------------------
// LayerNorm over last dim (512) -> f16 output for WMMA consumption
// ---------------------------------------------------------------------------
__global__ __launch_bounds__(256)
void layernorm_to_f16(const float* __restrict__ x,
                      const float* __restrict__ w,
                      const float* __restrict__ b,
                      _Float16* __restrict__ out) {
    __shared__ float s_sum[256];
    __shared__ float s_sq[256];
    const int m = blockIdx.x;
    const int t = threadIdx.x;
    const float* xr = x + (size_t)m * D_MODEL;
    float v0 = xr[t];
    float v1 = xr[t + 256];
    s_sum[t] = v0 + v1;
    s_sq[t]  = v0 * v0 + v1 * v1;
    __syncthreads();
    for (int off = 128; off > 0; off >>= 1) {
        if (t < off) { s_sum[t] += s_sum[t + off]; s_sq[t] += s_sq[t + off]; }
        __syncthreads();
    }
    const float mu  = s_sum[0] * (1.0f / D_MODEL);
    const float var = s_sq[0] * (1.0f / D_MODEL) - mu * mu;
    const float rstd = rsqrtf(var + 1e-5f);
    _Float16* orow = out + (size_t)m * D_MODEL;
    orow[t]       = (_Float16)((v0 - mu) * rstd * w[t] + b[t]);
    orow[t + 256] = (_Float16)((v1 - mu) * rstd * w[t + 256] + b[t + 256]);
}

// ---------------------------------------------------------------------------
// tile staging helpers: 128 rows x 32 halfs per tile = 512 x 16B chunks,
// 2 chunks per thread (256 threads)
// ---------------------------------------------------------------------------
__device__ __forceinline__ void load_tile_regs(const _Float16* __restrict__ base,
                                               int ldk, int row0, int k0,
                                               int rowMax, uint4 v[2], int tid) {
    #pragma unroll
    for (int i = 0; i < 2; ++i) {
        const int c   = tid + i * 256;      // 0..511
        const int r   = c >> 2;             // 0..127
        const int col = (c & 3) * 8;        // half offset 0,8,16,24
        const int row = row0 + r;
        uint4 t = {0u, 0u, 0u, 0u};
        if (row < rowMax) t = *(const uint4*)(base + (size_t)row * ldk + k0 + col);
        v[i] = t;
    }
}

__device__ __forceinline__ void store_tile_lds(unsigned* __restrict__ buf,
                                               const uint4 v[2], int tid) {
    #pragma unroll
    for (int i = 0; i < 2; ++i) {
        const int c   = tid + i * 256;
        const int r   = c >> 2;
        const int col = (c & 3) * 8;
        unsigned* dst = &buf[r * LDS_STRIDE + (col >> 1)];
        dst[0] = v[i].x; dst[1] = v[i].y; dst[2] = v[i].z; dst[3] = v[i].w;
    }
}

__device__ __forceinline__ void gather_frag(Frag& f, const unsigned* __restrict__ row) {
    f.u[0] = row[0]; f.u[1] = row[1]; f.u[2] = row[2];  f.u[3] = row[3];
    f.u[4] = row[8]; f.u[5] = row[9]; f.u[6] = row[10]; f.u[7] = row[11];
}

// ---------------------------------------------------------------------------
// WMMA GEMM: D[M,N] = A[M,K] * Bw[N,K]^T   (f16 in, f32 accumulate/out)
// 256 threads = 8 waves (4 M x 2 N); wave tile 32x64 = 8 WMMAs per K-step.
// LDS double-buffered: one barrier per K-step, next-tile global loads
// issued before the WMMA burst.
// epilogue: 0 = plain store, 1 = softplus(acc + bias[n]), 2 = resid + acc
// ---------------------------------------------------------------------------
__global__ __launch_bounds__(256)
void gemm_f16_wmma(const _Float16* __restrict__ A,
                   const _Float16* __restrict__ Bw,
                   float* __restrict__ D,
                   int M, int N, int K, int mode,
                   const float* __restrict__ bias,
                   const float* __restrict__ resid) {
    __shared__ unsigned As[2][BM * LDS_STRIDE];
    __shared__ unsigned Bs[2][BN * LDS_STRIDE];

    const int tid  = threadIdx.x;
    const int lane = tid & 31;
    const int wave = tid >> 5;
    const int wm   = wave >> 1;        // 0..3  -> M offset wm*32
    const int wn   = wave & 1;         // 0..1  -> N offset wn*64
    const int blockM = blockIdx.y * BM;
    const int blockN = blockIdx.x * BN;

    const int m16 = lane & 15;
    const int kh  = lane >> 4;

    v8f acc[2][4] = {};

    uint4 aReg[2], bReg[2];
    load_tile_regs(A,  K, blockM, 0, M, aReg, tid);
    load_tile_regs(Bw, K, blockN, 0, N, bReg, tid);
    store_tile_lds(As[0], aReg, tid);
    store_tile_lds(Bs[0], bReg, tid);

    const int nsteps = K >> 5;
    for (int s = 0; s < nsteps; ++s) {
        __syncthreads();                     // buffer `cur` ready / other buffer free
        const int cur = s & 1;

        if (s + 1 < nsteps) {                // issue next-tile loads early
            load_tile_regs(A,  K, blockM, (s + 1) * 32, M, aReg, tid);
            load_tile_regs(Bw, K, blockN, (s + 1) * 32, N, bReg, tid);
        }

        Frag af[2];
        #pragma unroll
        for (int i = 0; i < 2; ++i)
            gather_frag(af[i], &As[cur][(wm * 32 + i * 16 + m16) * LDS_STRIDE + kh * 4]);

        #pragma unroll
        for (int j = 0; j < 4; ++j) {
            Frag bf;
            gather_frag(bf, &Bs[cur][(wn * 64 + j * 16 + m16) * LDS_STRIDE + kh * 4]);
            #pragma unroll
            for (int i = 0; i < 2; ++i)
                acc[i][j] = __builtin_amdgcn_wmma_f32_16x16x32_f16(
                    false, af[i].h, false, bf.h, (short)0, acc[i][j], false, false);
        }

        if (s + 1 < nsteps) {                // fill the other buffer (free since top barrier)
            store_tile_lds(As[1 - cur], aReg, tid);
            store_tile_lds(Bs[1 - cur], bReg, tid);
        }
    }

    // store: VGPR r -> M = r + kh*8, N = m16 (per ISA C/D layout)
    #pragma unroll
    for (int i = 0; i < 2; ++i) {
        #pragma unroll
        for (int j = 0; j < 4; ++j) {
            const int gn = blockN + wn * 64 + j * 16 + m16;
            if (gn >= N) continue;
            #pragma unroll
            for (int r = 0; r < 8; ++r) {
                const int gm = blockM + wm * 32 + i * 16 + kh * 8 + r;
                float v = acc[i][j][r];
                const size_t o = (size_t)gm * N + gn;
                if (mode == 1) {
                    v += bias[gn];
                    v = (v > 20.0f) ? v : log1pf(__expf(v));   // softplus
                } else if (mode == 2) {
                    v += resid[o];                              // residual add
                }
                D[o] = v;
            }
        }
    }
}

// ---------------------------------------------------------------------------
// depthwise causal conv (k=4) + SiLU ; reads xin half of xz [tok, 2048]
// ---------------------------------------------------------------------------
__global__ void conv_silu(const float* __restrict__ xz,
                          const float* __restrict__ cw,
                          const float* __restrict__ cb,
                          float* __restrict__ xin,
                          _Float16* __restrict__ xin16) {
    const long long idx = (long long)blockIdx.x * 256 + threadIdx.x;
    if (idx >= (long long)NTOK * D_INNER) return;
    const int d  = (int)(idx & (D_INNER - 1));
    const int m  = (int)(idx >> 10);
    const int l  = m & (SEQ_L - 1);
    const int bb = m >> 11;
    float acc = cb[d];
    #pragma unroll
    for (int j = 0; j < 4; ++j) {
        const int ls = l - 3 + j;
        if (ls >= 0)
            acc += cw[d * 4 + j] * xz[((size_t)(bb * SEQ_L + ls)) * (2 * D_INNER) + d];
    }
    const float s = fast_silu(acc);
    xin[idx]   = s;
    xin16[idx] = (_Float16)s;
}

// extract dt (first 32 cols of xdbl) as f16 for the dt_proj GEMM
__global__ void extract_dt16(const float* __restrict__ xdbl,
                             _Float16* __restrict__ dt16) {
    const int idx = blockIdx.x * 256 + threadIdx.x;
    if (idx >= NTOK * DT_RANK) return;
    const int m = idx >> 5, c = idx & 31;
    dt16[idx] = (_Float16)xdbl[(size_t)m * 64 + c];
}

// ---------------------------------------------------------------------------
// selective scan: one thread per (batch, channel); 16 states in registers,
// B_t/C_t double-buffered through LDS -> ONE barrier per timestep.
// ---------------------------------------------------------------------------
__global__ __launch_bounds__(256)
void selective_scan(const float* __restrict__ dtsp,
                    const float* __restrict__ xin,
                    const float* __restrict__ xdbl,
                    const float* __restrict__ A_log,
                    const float* __restrict__ Dp,
                    float* __restrict__ ybuf) {
    __shared__ float sBC[2][32];
    const int t  = threadIdx.x;
    const int bb = blockIdx.x >> 2;
    const int d  = ((blockIdx.x & 3) << 8) + t;

    float Arow[D_STATE], h[D_STATE];
    #pragma unroll
    for (int s = 0; s < D_STATE; ++s) {
        Arow[s] = -__expf(A_log[d * D_STATE + s]);
        h[s] = 0.0f;
    }
    const float dpar = Dp[d];
    const size_t mBase = (size_t)bb * SEQ_L;

    for (int l = 0; l < SEQ_L; ++l) {
        const size_t m = mBase + l;
        const int buf = l & 1;
        if (t < 32) sBC[buf][t] = xdbl[m * 64 + DT_RANK + t];
        __syncthreads();                     // also fences prior readers of `buf`
        const float dt = dtsp[m * D_INNER + d];
        const float x  = xin[m * D_INNER + d];
        const float dtx = dt * x;
        float y = 0.0f;
        #pragma unroll
        for (int s = 0; s < D_STATE; ++s) {
            const float dA = __expf(dt * Arow[s]);
            h[s] = dA * h[s] + dtx * sBC[buf][s];
            y += h[s] * sBC[buf][D_STATE + s];
        }
        ybuf[m * D_INNER + d] = y + x * dpar;
    }
}

// gate: y *= silu(z), emit f16 for out_proj GEMM
__global__ void gate_silu(const float* __restrict__ xz,
                          const float* __restrict__ ybuf,
                          _Float16* __restrict__ y16) {
    const long long idx = (long long)blockIdx.x * 256 + threadIdx.x;
    if (idx >= (long long)NTOK * D_INNER) return;
    const int d = (int)(idx & (D_INNER - 1));
    const int m = (int)(idx >> 10);
    const float z = xz[(size_t)m * (2 * D_INNER) + D_INNER + d];
    y16[idx] = (_Float16)(ybuf[idx] * fast_silu(z));
}

// ---------------------------------------------------------------------------
extern "C" void kernel_launch(void* const* d_in, const int* in_sizes, int n_in,
                              void* d_out, int out_size, void* d_ws, size_t ws_size,
                              hipStream_t stream) {
    const float* hidden     = (const float*)d_in[0];
    const float* norm_w     = (const float*)d_in[1];
    const float* norm_b     = (const float*)d_in[2];
    const float* in_proj_w  = (const float*)d_in[3];   // [2048, 512]
    const float* conv_w     = (const float*)d_in[4];   // [1024, 1, 4]
    const float* conv_b     = (const float*)d_in[5];
    const float* x_proj_w   = (const float*)d_in[6];   // [64, 1024]
    const float* dt_proj_w  = (const float*)d_in[7];   // [1024, 32]
    const float* dt_proj_b  = (const float*)d_in[8];
    const float* A_log      = (const float*)d_in[9];
    const float* D_param    = (const float*)d_in[10];
    const float* out_proj_w = (const float*)d_in[11];  // [512, 1024]
    float* out = (float*)d_out;

    // ---- workspace layout (all offsets 256B-aligned) ----
    char* ws = (char*)d_ws;
    size_t off = 0;
    _Float16* xn16   = (_Float16*)(ws + off); off += (size_t)NTOK * D_MODEL * 2;
    _Float16* win16  = (_Float16*)(ws + off); off += (size_t)2 * D_INNER * D_MODEL * 2;
    _Float16* wxp16  = (_Float16*)(ws + off); off += (size_t)64 * D_INNER * 2;
    _Float16* wdt16  = (_Float16*)(ws + off); off += (size_t)D_INNER * DT_RANK * 2;
    _Float16* wout16 = (_Float16*)(ws + off); off += (size_t)D_MODEL * D_INNER * 2;
    float*    xz     = (float*)   (ws + off); off += (size_t)NTOK * 2 * D_INNER * 4;
    float*    xin    = (float*)   (ws + off); off += (size_t)NTOK * D_INNER * 4;
    _Float16* xin16  = (_Float16*)(ws + off); off += (size_t)NTOK * D_INNER * 2;
    float*    xdbl   = (float*)   (ws + off); off += (size_t)NTOK * 64 * 4;
    float*    dtsp   = (float*)   (ws + off); off += (size_t)NTOK * D_INNER * 4;
    _Float16* dt16   = (_Float16*)(ws + off); off += (size_t)NTOK * DT_RANK * 2;
    float*    ybuf   = (float*)   (ws + off); off += (size_t)NTOK * D_INNER * 4;
    _Float16* y16    = (_Float16*)(ws + off); off += (size_t)NTOK * D_INNER * 2;

    // ---- weight conversions to f16 ----
    {
        int n = 2 * D_INNER * D_MODEL;
        cvt_f32_to_f16<<<(n + 255) / 256, 256, 0, stream>>>(in_proj_w, win16, n);
        n = 64 * D_INNER;
        cvt_f32_to_f16<<<(n + 255) / 256, 256, 0, stream>>>(x_proj_w, wxp16, n);
        n = D_INNER * DT_RANK;
        cvt_f32_to_f16<<<(n + 255) / 256, 256, 0, stream>>>(dt_proj_w, wdt16, n);
        n = D_MODEL * D_INNER;
        cvt_f32_to_f16<<<(n + 255) / 256, 256, 0, stream>>>(out_proj_w, wout16, n);
    }

    // ---- 1. LayerNorm -> xn16 ----
    layernorm_to_f16<<<NTOK, 256, 0, stream>>>(hidden, norm_w, norm_b, xn16);

    // ---- 2. in_proj GEMM: [16384,512] x [2048,512]^T -> xz ----
    {
        dim3 g((2 * D_INNER + BN - 1) / BN, NTOK / BM);
        gemm_f16_wmma<<<g, 256, 0, stream>>>(xn16, win16, xz,
                                             NTOK, 2 * D_INNER, D_MODEL,
                                             0, nullptr, nullptr);
    }

    // ---- 3. depthwise causal conv + SiLU -> xin / xin16 ----
    {
        long long n = (long long)NTOK * D_INNER;
        conv_silu<<<(unsigned)((n + 255) / 256), 256, 0, stream>>>(xz, conv_w, conv_b,
                                                                   xin, xin16);
    }

    // ---- 4. x_proj GEMM: [16384,1024] x [64,1024]^T -> xdbl ----
    {
        dim3 g((64 + BN - 1) / BN, NTOK / BM);
        gemm_f16_wmma<<<g, 256, 0, stream>>>(xin16, wxp16, xdbl,
                                             NTOK, 64, D_INNER,
                                             0, nullptr, nullptr);
    }

    // ---- 5. dt_proj GEMM + bias + softplus -> dtsp ----
    {
        int n = NTOK * DT_RANK;
        extract_dt16<<<(n + 255) / 256, 256, 0, stream>>>(xdbl, dt16);
        dim3 g((D_INNER + BN - 1) / BN, NTOK / BM);
        gemm_f16_wmma<<<g, 256, 0, stream>>>(dt16, wdt16, dtsp,
                                             NTOK, D_INNER, DT_RANK,
                                             1, dt_proj_b, nullptr);
    }

    // ---- 6. selective scan -> ybuf (= y + xin*D) ----
    selective_scan<<<SEQ_B * (D_INNER / 256), 256, 0, stream>>>(
        dtsp, xin, xdbl, A_log, D_param, ybuf);

    // ---- 7. gate: y *= silu(z) -> y16 ----
    {
        long long n = (long long)NTOK * D_INNER;
        gate_silu<<<(unsigned)((n + 255) / 256), 256, 0, stream>>>(xz, ybuf, y16);
    }

    // ---- 8. out_proj GEMM + residual -> d_out ----
    {
        dim3 g((D_MODEL + BN - 1) / BN, NTOK / BM);
        gemm_f16_wmma<<<g, 256, 0, stream>>>(y16, wout16, out,
                                             NTOK, D_MODEL, D_INNER,
                                             2, nullptr, hidden);
    }
}
